// ESALayer_66752381714660
// MI455X (gfx1250) — matrix operand order repaired and checked by reading, not compiled, not run on hardware
//
#include <hip/hip_runtime.h>
#include <hip/hip_bf16.h>

// ---------------------------------------------------------------------------
// Types for CDNA5 WMMA (wave32): bf16 16x16x32, fp32 accumulate
// ---------------------------------------------------------------------------
typedef __bf16 bf16_t;
typedef __attribute__((ext_vector_type(16))) __bf16 v16bf;
typedef __attribute__((ext_vector_type(8)))  __bf16 v8bf;
typedef __attribute__((ext_vector_type(8)))  float  v8f;

__device__ __forceinline__ bf16_t f2bf(float f) {
  // round-to-nearest-even fp32 -> bf16
  unsigned u = __float_as_uint(f);
  u += 0x7FFFu + ((u >> 16) & 1u);
  unsigned short h = (unsigned short)(u >> 16);
  bf16_t r;
  __builtin_memcpy(&r, &h, sizeof(r));
  return r;
}

union Frag {
  v16bf v;
  v8bf  h[2];
};

// CDNA5 async copy: global -> LDS, 16B per lane, tracked by ASYNCcnt.
// Inline asm (not the clang builtin) so it is portable across toolchains.
__device__ __forceinline__ void async_load_b128(unsigned lds_byte_off,
                                                const void* gptr) {
  asm volatile("global_load_async_to_lds_b128 %0, %1, off"
               :
               : "v"(lds_byte_off), "v"(gptr)
               : "memory");
}

template <int N>
__device__ __forceinline__ void wait_asynccnt() {
  asm volatile("s_wait_asynccnt %0" ::"n"(N) : "memory");
}

// ---------------------------------------------------------------------------
// Generic bf16 WMMA GEMM:  C[M,N] = scale * (A[M,K] @ Bt[N,K]^T) + bias[N]
//   A  : row-major bf16, leading dim K (wave-private rows, direct global)
//   Bt : row-major bf16 holding B^T (Bt[n][k] = B[k][n]), leading dim K
//        -> staged through double-buffered LDS with async-to-LDS copies
// MODE 0: store bf16 row-major (ldc = N)
// MODE 1: store bf16 transposed per batch-of-rows:
//           C[(row/rpb)*sC + col*rpb + row%rpb]   (used to produce V^T)
// MODE 2: store fp32 row-major (ldc = N)
// Batch via gridDim.z with element strides sA, sB, sC.
// Block = 256 threads = 8 waves; block tile 128(M) x 64(N); wave tile 16x64.
// K-step 64, B tile 64x64 bf16 in LDS, pitch 72 elems (144B, 9x16B -> no
// 16B-granule bank conflicts), double buffered: 2 x 9216 B dynamic LDS.
// All fragment loads for a stage are hoisted ahead of the 8 WMMAs so the
// scheduler can overlap ds/global latency with matrix ops (partial waits).
// Requires K % 64 == 0, M % 128 == 0, N % 64 == 0.
// ---------------------------------------------------------------------------
#define BPITCH 144   // bytes per B-tile row in LDS (64*2 + 16 pad)
#define BBUF   9216  // bytes per B-tile buffer (64 rows * 144)

template <int MODE>
__global__ __launch_bounds__(256) void gemm_bf16_wmma(
    const bf16_t* __restrict__ A, const bf16_t* __restrict__ Bt,
    const float* __restrict__ bias, float scale, void* __restrict__ Cv,
    int M, int N, int K, long sA, long sB, long sC, int rowsPerBatchC) {
  extern __shared__ char smem[];  // 2 * BBUF bytes, starts at LDS offset 0
  const int tid  = threadIdx.x;
  const int lane = tid & 31;
  const int wave = tid >> 5;
  const long bz  = blockIdx.z;
  A  += bz * sA;
  Bt += bz * sB;

  const int m0 = blockIdx.y * 128 + wave * 16;
  const int n0 = blockIdx.x * 64;

  // --- async B-tile fill: 64 rows x 128B = 512 x 16B segs, 2 per thread ---
  const int s0r = tid >> 3,          s0k = tid & 7;          // seg j = 0
  const int s1r = (256 + tid) >> 3,  s1k = tid & 7;          // seg j = 1
  const bf16_t* gB0 = Bt + (long)(n0 + s0r) * K + s0k * 8;
  const bf16_t* gB1 = Bt + (long)(n0 + s1r) * K + s1k * 8;
  const unsigned lB0 = (unsigned)(s0r * BPITCH + s0k * 16);
  const unsigned lB1 = (unsigned)(s1r * BPITCH + s1k * 16);

  // --- A fragment addressing (16x32 bf16): lane%16 = row; lanes 0-15 hold
  // K = {k..k+7, k+16..k+23}, lanes 16-31 hold K = {+8..+15, +24..+31}. ---
  const bf16_t* arow = A + (long)(m0 + (lane & 15)) * K + ((lane >> 4) << 3);

  // --- B fragment addressing (32x16 bf16): lane%16 = col; lanes 0-15 hold
  // K = {k..k+7, k+16..k+23}, lanes 16-31 hold K = {+8..+15, +24..+31}. ---
  const int bcol  = lane & 15;
  const int bkoff = (lane >> 4) << 3;  // elements

  v8f acc[4] = {};
  const int S = K >> 6;  // stages of 64 K

  // prologue: stage 0 into buffer 0
  async_load_b128(lB0, gB0);
  async_load_b128(lB1, gB1);

  for (int s = 0; s < S; ++s) {
    if (s + 1 < S) {
      const long kg = (long)(s + 1) << 6;
      const unsigned boff = (unsigned)(((s + 1) & 1) * BBUF);
      async_load_b128(lB0 + boff, gB0 + kg);
      async_load_b128(lB1 + boff, gB1 + kg);
      wait_asynccnt<2>();  // stage s complete; stage s+1 still in flight
    } else {
      wait_asynccnt<0>();
    }
    __syncthreads();

    const char* bufB = smem + (s & 1) * BBUF;
    const long  k0g  = (long)s << 6;

    // Hoist all fragment loads for this stage (2 A from global, 8 B from
    // LDS) so the 8 WMMAs below only need partial dscnt/loadcnt waits.
    Frag a0, a1, b0[4], b1[4];
    a0.h[0] = *(const v8bf*)(arow + k0g);
    a0.h[1] = *(const v8bf*)(arow + k0g + 16);
    a1.h[0] = *(const v8bf*)(arow + k0g + 32);
    a1.h[1] = *(const v8bf*)(arow + k0g + 48);
#pragma unroll
    for (int t = 0; t < 4; ++t) {
      const char* bp = bufB + (t * 16 + bcol) * BPITCH + bkoff * 2;
      b0[t].h[0] = ((const v8bf*)bp)[0];
      b0[t].h[1] = ((const v8bf*)bp)[2];
    }
#pragma unroll
    for (int t = 0; t < 4; ++t) {
      const char* bp = bufB + (t * 16 + bcol) * BPITCH + (32 + bkoff) * 2;
      b1[t].h[0] = ((const v8bf*)bp)[0];
      b1[t].h[1] = ((const v8bf*)bp)[2];
    }
#pragma unroll
    for (int t = 0; t < 4; ++t) {
      acc[t] = __builtin_amdgcn_wmma_f32_16x16x32_bf16(
          false, a0.v, false, b0[t].v, (short)0, acc[t], false, false);
    }
#pragma unroll
    for (int t = 0; t < 4; ++t) {
      acc[t] = __builtin_amdgcn_wmma_f32_16x16x32_bf16(
          false, a1.v, false, b1[t].v, (short)0, acc[t], false, false);
    }
    __syncthreads();  // protect buffer (s&1) before it is refilled
  }

  // C/D layout: VGPR g holds M = m0 + g (lanes 0-15) or m0 + 8 + g (16-31),
  // N = n0 + lane%16.
  const int rbase = m0 + ((lane >> 4) << 3);
  const int cbase = n0 + (lane & 15);
#pragma unroll
  for (int t = 0; t < 4; ++t) {
    const int col = cbase + t * 16;
    const float bv = bias ? bias[col] : 0.0f;
#pragma unroll
    for (int g = 0; g < 8; ++g) {
      const int row = rbase + g;
      const float val = acc[t][g] * scale + bv;
      if (MODE == 2) {
        ((float*)Cv)[bz * sC + (long)row * N + col] = val;
      } else if (MODE == 0) {
        ((bf16_t*)Cv)[bz * sC + (long)row * N + col] = f2bf(val);
      } else {
        const int b2 = row / rowsPerBatchC;
        const int r2 = row - b2 * rowsPerBatchC;
        ((bf16_t*)Cv)[(long)b2 * sC + (long)col * rowsPerBatchC + r2] =
            f2bf(val);
      }
    }
  }
}

// ---------------------------------------------------------------------------
// LayerNorm over last dim (d = 1024), fp32 in -> bf16 out. One row per block.
// ---------------------------------------------------------------------------
__global__ __launch_bounds__(256) void layernorm_to_bf16(
    const float* __restrict__ x, const float* __restrict__ gamma,
    const float* __restrict__ beta, bf16_t* __restrict__ out, int d) {
  const int tid = threadIdx.x;
  const long row = blockIdx.x;
  const float4 v = ((const float4*)(x + row * d))[tid];

  __shared__ float r1[256], r2[256];
  r1[tid] = v.x + v.y + v.z + v.w;
  r2[tid] = v.x * v.x + v.y * v.y + v.z * v.z + v.w * v.w;
  __syncthreads();
  for (int s = 128; s > 0; s >>= 1) {
    if (tid < s) {
      r1[tid] += r1[tid + s];
      r2[tid] += r2[tid + s];
    }
    __syncthreads();
  }
  const float mu  = r1[0] / (float)d;
  const float var = r2[0] / (float)d - mu * mu;
  const float inv = rsqrtf(var + 1e-5f);

  const float4 g = ((const float4*)gamma)[tid];
  const float4 b = ((const float4*)beta)[tid];
  union { bf16_t e[4]; unsigned long long u; } pk;
  pk.e[0] = f2bf((v.x - mu) * inv * g.x + b.x);
  pk.e[1] = f2bf((v.y - mu) * inv * g.y + b.y);
  pk.e[2] = f2bf((v.z - mu) * inv * g.z + b.z);
  pk.e[3] = f2bf((v.w - mu) * inv * g.w + b.w);
  ((unsigned long long*)(out + row * d))[tid] = pk.u;
}

// ---------------------------------------------------------------------------
// Row softmax (m = 1024), fp32 scores in -> bf16 probabilities out.
// ---------------------------------------------------------------------------
__global__ __launch_bounds__(256) void softmax_to_bf16(
    const float* __restrict__ S, bf16_t* __restrict__ P, int m) {
  const int tid = threadIdx.x;
  const long row = blockIdx.x;
  const float4 v = ((const float4*)(S + row * m))[tid];

  __shared__ float red[256];
  float mx = fmaxf(fmaxf(v.x, v.y), fmaxf(v.z, v.w));
  red[tid] = mx;
  __syncthreads();
  for (int s = 128; s > 0; s >>= 1) {
    if (tid < s) red[tid] = fmaxf(red[tid], red[tid + s]);
    __syncthreads();
  }
  mx = red[0];
  __syncthreads();

  const float e0 = __expf(v.x - mx), e1 = __expf(v.y - mx);
  const float e2 = __expf(v.z - mx), e3 = __expf(v.w - mx);
  red[tid] = e0 + e1 + e2 + e3;
  __syncthreads();
  for (int s = 128; s > 0; s >>= 1) {
    if (tid < s) red[tid] += red[tid + s];
    __syncthreads();
  }
  const float inv = 1.0f / red[0];

  union { bf16_t e[4]; unsigned long long u; } pk;
  pk.e[0] = f2bf(e0 * inv);
  pk.e[1] = f2bf(e1 * inv);
  pk.e[2] = f2bf(e2 * inv);
  pk.e[3] = f2bf(e3 * inv);
  ((unsigned long long*)(P + row * m))[tid] = pk.u;
}

// ---------------------------------------------------------------------------
// W[Kd,Nd] fp32 (row-major) -> Wt[Nd,Kd] bf16 (row-major), coalesced writes.
// ---------------------------------------------------------------------------
__global__ __launch_bounds__(256) void transpose_to_bf16(
    const float* __restrict__ W, bf16_t* __restrict__ Wt, int Kd, int Nd) {
  const long idx = (long)blockIdx.x * 256 + threadIdx.x;
  if (idx >= (long)Kd * Nd) return;
  const int k = (int)(idx % Kd);
  const int n = (int)(idx / Kd);
  Wt[idx] = f2bf(W[(long)k * Nd + n]);
}

// ---------------------------------------------------------------------------
// Launch: LN -> Q/K/V GEMMs -> scores -> softmax -> attn@V -> out proj
// ---------------------------------------------------------------------------
extern "C" void kernel_launch(void* const* d_in, const int* in_sizes, int n_in,
                              void* d_out, int out_size, void* d_ws,
                              size_t ws_size, hipStream_t stream) {
  (void)in_sizes; (void)n_in; (void)out_size; (void)ws_size;

  const float* x    = (const float*)d_in[0];
  const float* ln_g = (const float*)d_in[1];
  const float* ln_b = (const float*)d_in[2];
  const float* Wq   = (const float*)d_in[3];
  const float* bq   = (const float*)d_in[4];
  const float* Wk   = (const float*)d_in[5];
  const float* bk   = (const float*)d_in[6];
  const float* Wv   = (const float*)d_in[7];
  const float* bv   = (const float*)d_in[8];
  const float* Wo   = (const float*)d_in[9];
  const float* bo   = (const float*)d_in[10];

  constexpr int D = 1024, DR = 4096;        // dim, dim*ratio
  constexpr int Btch = 4, Ntok = 4096, Mtok = 1024;
  constexpr long ROWS = (long)Btch * Ntok;  // 16384 query rows
  constexpr long KVR  = (long)Btch * Mtok;  // 4096 kv rows
  constexpr unsigned SMEM = 2 * BBUF;       // double-buffered B tile

  char* w = (char*)d_ws;
  auto take = [&](size_t bytes) {
    char* p = w;
    w += (bytes + 255) & ~(size_t)255;
    return p;
  };
  bf16_t* WqT = (bf16_t*)take((size_t)D * D * 2);
  bf16_t* WkT = (bf16_t*)take((size_t)D * DR * 2);
  bf16_t* WvT = (bf16_t*)take((size_t)D * DR * 2);
  bf16_t* WoT = (bf16_t*)take((size_t)D * D * 2);
  bf16_t* xn  = (bf16_t*)take((size_t)ROWS * D * 2);
  bf16_t* qb  = (bf16_t*)take((size_t)ROWS * D * 2);
  bf16_t* kb  = (bf16_t*)take((size_t)KVR * D * 2);
  bf16_t* vt  = (bf16_t*)take((size_t)Btch * D * Mtok * 2);  // V^T per batch
  float*  sc  = (float*)take((size_t)ROWS * Mtok * 4);       // fp32 scores
  bf16_t* attn = xn;           // xn dead after Q/K/V GEMMs
  bf16_t* oatt = (bf16_t*)sc;  // scores dead after softmax

  // Weights -> transposed bf16 (Bt layout: [N][K])
  {
    const long n1 = (long)D * D;
    const long n2 = (long)D * DR;
    transpose_to_bf16<<<dim3((unsigned)((n1 + 255) / 256)), 256, 0, stream>>>(
        Wq, WqT, D, D);
    transpose_to_bf16<<<dim3((unsigned)((n2 + 255) / 256)), 256, 0, stream>>>(
        Wk, WkT, DR, D);
    transpose_to_bf16<<<dim3((unsigned)((n2 + 255) / 256)), 256, 0, stream>>>(
        Wv, WvT, DR, D);
    transpose_to_bf16<<<dim3((unsigned)((n1 + 255) / 256)), 256, 0, stream>>>(
        Wo, WoT, D, D);
  }

  // LayerNorm -> bf16
  layernorm_to_bf16<<<dim3((unsigned)ROWS), 256, 0, stream>>>(x, ln_g, ln_b,
                                                              xn, D);

  // Q = xn @ Wq + bq                        (16384 x 1024, K = 1024)
  gemm_bf16_wmma<0><<<dim3(D / 64, (unsigned)(ROWS / 128), 1), 256, SMEM,
                      stream>>>(xn, WqT, bq, 1.0f, qb, (int)ROWS, D, D, 0, 0,
                                0, 0);
  // K = xr @ Wk + bk   (xr = xn viewed 4096 x 4096)
  gemm_bf16_wmma<0><<<dim3(D / 64, (unsigned)(KVR / 128), 1), 256, SMEM,
                      stream>>>(xn, WkT, bk, 1.0f, kb, (int)KVR, D, DR, 0, 0,
                                0, 0);
  // V = xr @ Wv + bv, stored transposed per batch: vt[b][d][m]
  gemm_bf16_wmma<1><<<dim3(D / 64, (unsigned)(KVR / 128), 1), 256, SMEM,
                      stream>>>(xn, WvT, bv, 1.0f, vt, (int)KVR, D, DR, 0, 0,
                                (long)D * Mtok, Mtok);

  // scores[b] = q[b] @ k[b]^T / 32   (fp32, k rows already serve as Bt)
  gemm_bf16_wmma<2><<<dim3(Mtok / 64, Ntok / 128, Btch), 256, SMEM, stream>>>(
      qb, kb, nullptr, 0.03125f, sc, Ntok, Mtok, D, (long)Ntok * D,
      (long)Mtok * D, (long)Ntok * Mtok, 0);

  // softmax over M -> bf16 attn
  softmax_to_bf16<<<dim3((unsigned)ROWS), 256, 0, stream>>>(sc, attn, Mtok);

  // out[b] = attn[b] @ v[b]    (Bt = vt[b])
  gemm_bf16_wmma<0><<<dim3(D / 64, Ntok / 128, Btch), 256, SMEM, stream>>>(
      attn, vt, nullptr, 1.0f, oatt, Ntok, D, Mtok, (long)Ntok * Mtok,
      (long)D * Mtok, (long)Ntok * D, 0);

  // result = out @ Wo + bo  (fp32 to d_out)
  gemm_bf16_wmma<2><<<dim3(D / 64, (unsigned)(ROWS / 128), 1), 256, SMEM,
                      stream>>>(oatt, WoT, bo, 1.0f, (float*)d_out, (int)ROWS,
                                D, D, 0, 0, 0, 0);
}
